// RTBlock_35562329211150
// MI455X (gfx1250) — matrix operand order repaired and checked by reading, not compiled, not run on hardware
//
#include <hip/hip_runtime.h>

// ---------------------------------------------------------------------------
// CDNA5 (gfx1250): bf16 WMMA GEMMs, LDS-staged 64(M)x64(N)x64(K) block tiles.
// Block = 128 threads (4 waves). Per staged tile: coalesced b128 global loads
// -> LDS (A row-major [m][k64], B transposed [n][k64]) -> ds_load_b128
// fragments -> 8 x v_wmma_f32_16x16x32_bf16 per wave per barrier pair.
// 3x3 conv: zero-padded image tile (4 x 34 x 64ch, channel-innermost) staged
// once per K-chunk and reused by all 9 taps.
// Algebraic simplification: conv1x1(upsample_nearest(x,4)) ==
// upsample_nearest(conv1x1(x),4) -> output convs run at 32x32.
// ---------------------------------------------------------------------------

typedef __attribute__((ext_vector_type(16))) __bf16 v16bf;
typedef __attribute__((ext_vector_type(8)))  __bf16 v8bf;
typedef __attribute__((ext_vector_type(8)))  float  v8f;

#define BB   8
#define CCH  256
#define ICH  128
#define NSP  1024    // 32*32 pooled spatial
#define RSP  16384   // 128*128 raw spatial

#define CAT16 0,1,2,3,4,5,6,7,8,9,10,11,12,13,14,15

// ---------------- small helper kernels ----------------

__global__ __launch_bounds__(256) void cvt_bf16_kernel(const float* __restrict__ s,
                                                       __bf16* __restrict__ d, int n) {
  int i = blockIdx.x * 256 + threadIdx.x;
  if (i < n) d[i] = (__bf16)s[i];
}

// repack (co,ci,3,3) f32 -> tap-major (9, co, ci) bf16 for implicit-GEMM conv
__global__ __launch_bounds__(256) void repack3_kernel(const float* __restrict__ W,
                                                      __bf16* __restrict__ Wt) {
  int i = blockIdx.x * 256 + threadIdx.x;     // 9*128*256
  if (i >= 9 * ICH * CCH) return;
  int tap = i / (ICH * CCH);
  int rem = i % (ICH * CCH);
  Wt[i] = (__bf16)W[(size_t)rem * 9 + tap];   // rem = co*CCH+ci
}

// 4x4 block-mean pool (128x128 -> 32x32), output bf16 (GEMM operand)
__global__ __launch_bounds__(256) void pool4_kernel(const float* __restrict__ X,
                                                    __bf16* __restrict__ P) {
  size_t idx = (size_t)blockIdx.x * 256 + threadIdx.x;   // BB*CCH*NSP total
  size_t bc = idx >> 10;
  int sp = (int)(idx & 1023);
  int py = sp >> 5, px = sp & 31;
  const float* src = X + bc * RSP + (size_t)(py * 4) * 128 + (size_t)(px * 4);
  float acc = 0.f;
#pragma unroll
  for (int i = 0; i < 4; ++i) {
    float4 q = *(const float4*)(src + i * 128);
    acc += q.x + q.y + q.z + q.w;
  }
  P[idx] = (__bf16)(acc * 0.0625f);
}

// nearest-neighbor x4 upsample (32x32 -> 128x128), f32 -> f32
__global__ __launch_bounds__(256) void upsample4_kernel(const float* __restrict__ Xs,
                                                        float* __restrict__ O) {
  size_t idx = (size_t)blockIdx.x * 256 + threadIdx.x;   // BB*CCH*RSP total
  size_t bc = idx >> 14;
  int sp = (int)(idx & 16383);
  int y = sp >> 7, x = sp & 127;
  O[idx] = Xs[bc * NSP + (size_t)((y >> 2) * 32 + (x >> 2))];
}

// row softmax over 1024 elements, f32 in -> bf16 out. blockIdx.x = row id.
__global__ __launch_bounds__(256) void softmax_kernel(const float* __restrict__ S,
                                                      __bf16* __restrict__ A) {
  __shared__ float red[256];
  size_t row = blockIdx.x;
  const float* s = S + row * 1024;
  __bf16* a = A + row * 1024;
  int t = threadIdx.x;
  float mx = -3.4e38f;
  for (int i = t; i < 1024; i += 256) mx = fmaxf(mx, s[i]);
  red[t] = mx; __syncthreads();
  for (int w = 128; w > 0; w >>= 1) { if (t < w) red[t] = fmaxf(red[t], red[t + w]); __syncthreads(); }
  mx = red[0]; __syncthreads();
  float sum = 0.f;
  for (int i = t; i < 1024; i += 256) sum += __expf(s[i] - mx);
  red[t] = sum; __syncthreads();
  for (int w = 128; w > 0; w >>= 1) { if (t < w) red[t] += red[t + w]; __syncthreads(); }
  float inv = 1.f / red[0];
  for (int i = t; i < 1024; i += 256) a[i] = (__bf16)(__expf(s[i] - mx) * inv);
}

// ---------------- device helpers for WMMA fragments ----------------
// A 16x32 layout: half 0 -> K 0..7 & 16..23 ; half 1 -> K 8..15 & 24..31
__device__ __forceinline__ v16bf frag_A(const __bf16* base, int half) {
  v8bf a0 = *(const v8bf*)(base + half * 8);
  v8bf a1 = *(const v8bf*)(base + 16 + half * 8);
  return __builtin_shufflevector(a0, a1, CAT16);
}
// B 32x16 layout: half 0 -> K 0..7 & 16..23 ; half 1 -> K 8..15 & 24..31
__device__ __forceinline__ v16bf frag_B(const __bf16* base, int half) {
  v8bf b0 = *(const v8bf*)(base + half * 8);
  v8bf b1 = *(const v8bf*)(base + 16 + half * 8);
  return __builtin_shufflevector(b0, b1, CAT16);
}

// ---------------- generic batched WMMA GEMM (64x64x64 tiles) ----------------
// out[b][m][n] = epilogue( sum_k A(b,m,k) * B(b,k,n) )
// A element (m,k) at A + b*Abs + m*sAm + k*sAk (bf16); sAk==1 -> vector path.
// B element (k,n) at B + b*Bbs + k*ldb + n     (bf16)
// Grid: (M/64, N/64, batch). Block: (32,4). K must be a multiple of 64.
__global__ __launch_bounds__(128) void gemm_bf16_kernel(
    const __bf16* __restrict__ A, long long Abs, long long sAm, long long sAk,
    const __bf16* __restrict__ Bm, long long Bbs, long long ldb,
    float* __restrict__ Cf, __bf16* __restrict__ Ch, long long Cbs, long long ldc,
    int K,
    const float* __restrict__ bias,
    const float* __restrict__ bng, const float* __restrict__ bnbe,
    const float* __restrict__ bnm, const float* __restrict__ bnv,
    const float* __restrict__ resid, long long Rbs, int relu)
{
  __shared__ __bf16 As[64 * 64];   // [m_local][k_local(64)]
  __shared__ __bf16 Bs[64 * 64];   // [n_local][k_local(64)] (transposed stage)

  int lane = threadIdx.x;
  int wv   = threadIdx.y;                     // wave id 0..3
  int t    = wv * 32 + lane;                  // 0..127
  int half = lane >> 4, r = lane & 15;
  int m0 = blockIdx.x * 64;
  int n0 = blockIdx.y * 64;
  int b  = blockIdx.z;

  const __bf16* Ab = A  + (size_t)b * Abs;
  const __bf16* Bb = Bm + (size_t)b * Bbs;

  v8f acc[4] = {};

  for (int k0 = 0; k0 < K; k0 += 64) {
    // ---- stage A: 64 rows x 64 k ----
    if (sAk == 1) {
      int row = t >> 1, seg = t & 1;          // each thread: 32 contiguous k
      const v8bf* g = (const v8bf*)(Ab + (size_t)(m0 + row) * sAm + (k0 + seg * 32));
      v8bf* s = (v8bf*)(As + row * 64 + seg * 32);
      s[0] = g[0]; s[1] = g[1]; s[2] = g[2]; s[3] = g[3];
    } else {  // m-contiguous (e.g. A = k^T): load rows of k, scatter-transpose
      int k = t & 63, mseg = t >> 6;          // each thread: 32 contiguous m
      const __bf16* gp = Ab + (size_t)(k0 + k) * sAk + (m0 + mseg * 32);
      const v8bf* g = (const v8bf*)gp;
#pragma unroll
      for (int q4 = 0; q4 < 4; ++q4) {
        v8bf v0 = g[q4];
#pragma unroll
        for (int e = 0; e < 8; ++e)
          As[(mseg * 32 + q4 * 8 + e) * 64 + k] = v0[e];
      }
    }
    // ---- stage B: 64 k x 64 n, transposed into Bs[n][k] ----
    {
      int k = t & 63, nseg = t >> 6;          // each thread: 32 contiguous n
      const v8bf* g = (const v8bf*)(Bb + (size_t)(k0 + k) * ldb + (n0 + nseg * 32));
#pragma unroll
      for (int q4 = 0; q4 < 4; ++q4) {
        v8bf v0 = g[q4];
#pragma unroll
        for (int e = 0; e < 8; ++e)
          Bs[(nseg * 32 + q4 * 8 + e) * 64 + k] = v0[e];
      }
    }
    __syncthreads();

#pragma unroll
    for (int ks = 0; ks < 2; ++ks) {
      v16bf fa = frag_A(As + (wv * 16 + r) * 64 + ks * 32, half);
#pragma unroll
      for (int j = 0; j < 4; ++j) {
        v16bf fb = frag_B(Bs + (j * 16 + r) * 64 + ks * 32, half);
        acc[j] = __builtin_amdgcn_wmma_f32_16x16x32_bf16(false, fa, false, fb,
                                                         (short)0, acc[j], false, false);
      }
    }
    __syncthreads();
  }

  // ---- epilogue ----
#pragma unroll
  for (int j = 0; j < 4; ++j) {
#pragma unroll
    for (int v = 0; v < 8; ++v) {
      int m = m0 + wv * 16 + half * 8 + v;
      int n = n0 + j * 16 + r;
      float xv = acc[j][v];
      if (bias) xv += bias[m];
      if (bng) {
        float sc = bng[m] * rsqrtf(bnv[m] + 1e-5f);
        xv = (xv - bnm[m]) * sc + bnbe[m];
      }
      if (relu) xv = fmaxf(xv, 0.f);
      if (resid) xv += resid[(size_t)b * Rbs + (size_t)m * ldc + n];
      size_t o = (size_t)b * Cbs + (size_t)m * ldc + n;
      if (Cf) Cf[o] = xv;
      if (Ch) Ch[o] = (__bf16)xv;
    }
  }
}

// ---------------- 3x3 conv as implicit WMMA GEMM ----------------
// Wt: tap-major (9,128,256) bf16; X: (B,256,32,32) bf16; Y: (B,128,32,32) bf16.
// A 64x64 LDS weight tile per tap; B fragments read from a zero-padded image
// tile IM[4 rows][34 cols][64 ch] (channel-innermost) staged once per K-chunk
// and shared by all 9 taps.
__global__ __launch_bounds__(128) void conv3_bf16_kernel(
    const __bf16* __restrict__ Wt, const __bf16* __restrict__ X,
    __bf16* __restrict__ Y)
{
  __shared__ __bf16 As[64 * 64];        // weights [co][ci(64)]
  __shared__ __bf16 IM[4 * 34 * 64];    // [row][col][ch]

  int lane = threadIdx.x;
  int wv   = threadIdx.y;
  int t    = wv * 32 + lane;
  int half = lane >> 4, r = lane & 15;
  int m0 = blockIdx.x * 64;             // out-channel tile base (M=128 -> 2)
  int n0 = blockIdx.y * 64;             // spatial tile base (2 image rows)
  int b  = blockIdx.z;
  int y0 = n0 >> 5;                     // first image row of this tile
  const __bf16* Xb = X + (size_t)b * CCH * NSP;

  v8f acc[4] = {};

  for (int k0 = 0; k0 < CCH; k0 += 64) {
    // ---- stage padded image tile: rows y0-1..y0+2, cols -1..32, 64 ch ----
    {
      int ch = t & 63, hp = t >> 6;     // hp selects half of the 34 columns
      const __bf16* Xc = Xb + (size_t)(k0 + ch) * NSP;
#pragma unroll
      for (int rrow = 0; rrow < 4; ++rrow) {
        int gy = y0 - 1 + rrow;
        bool yok = (unsigned)gy < 32u;
#pragma unroll
        for (int cc = 0; cc < 17; ++cc) {
          int col = hp * 17 + cc;
          int gx = col - 1;
          bool inb = yok && ((unsigned)gx < 32u);
          int src = inb ? (gy * 32 + gx) : 0;   // clamp address, select value
          __bf16 vv = Xc[src];
          IM[(rrow * 34 + col) * 64 + ch] = inb ? vv : (__bf16)0.0f;
        }
      }
    }
    for (int tap = 0; tap < 9; ++tap) {
      int dy = tap / 3 - 1, dx = tap % 3 - 1;
      { // A stage (contiguous K): 64 co x 64 ci
        const __bf16* Wtap = Wt + (size_t)tap * ICH * CCH;
        int row = t >> 1, seg = t & 1;
        const v8bf* g = (const v8bf*)(Wtap + (size_t)(m0 + row) * CCH + (k0 + seg * 32));
        v8bf* s = (v8bf*)(As + row * 64 + seg * 32);
        s[0] = g[0]; s[1] = g[1]; s[2] = g[2]; s[3] = g[3];
      }
      __syncthreads();

#pragma unroll
      for (int ks = 0; ks < 2; ++ks) {
        v16bf fa = frag_A(As + (wv * 16 + r) * 64 + ks * 32, half);
#pragma unroll
        for (int j = 0; j < 4; ++j) {
          int n = n0 + j * 16 + r;
          int rrow = (n >> 5) - y0 + dy + 1;    // 0..3
          int col  = (n & 31) + dx + 1;         // 0..33
          v16bf fb = frag_B(IM + (rrow * 34 + col) * 64 + ks * 32, half);
          acc[j] = __builtin_amdgcn_wmma_f32_16x16x32_bf16(false, fa, false, fb,
                                                           (short)0, acc[j], false, false);
        }
      }
      __syncthreads();
    }
  }

#pragma unroll
  for (int j = 0; j < 4; ++j) {
#pragma unroll
    for (int v = 0; v < 8; ++v) {
      int m = m0 + wv * 16 + half * 8 + v;
      int n = n0 + j * 16 + r;
      Y[(size_t)b * ICH * NSP + (size_t)m * NSP + n] = (__bf16)acc[j][v];
    }
  }
}

// ---------------------------------------------------------------------------

extern "C" void kernel_launch(void* const* d_in, const int* in_sizes, int n_in,
                              void* d_out, int out_size, void* d_ws, size_t ws_size,
                              hipStream_t stream) {
  (void)in_sizes; (void)n_in; (void)out_size; (void)ws_size;

  // ---- inputs (setup_inputs order) ----
  const float* x      = (const float*)d_in[0];
  const float* extra  = (const float*)d_in[1];
  const float* am0_w  = (const float*)d_in[2];
  const float* am0_b  = (const float*)d_in[3];
  const float* am0_g  = (const float*)d_in[4];
  const float* am0_be = (const float*)d_in[5];
  const float* am0_m  = (const float*)d_in[6];
  const float* am0_v  = (const float*)d_in[7];
  const float* am1_w  = (const float*)d_in[8];
  const float* am1_b  = (const float*)d_in[9];
  const float* am1_g  = (const float*)d_in[10];
  const float* am1_be = (const float*)d_in[11];
  const float* am1_m  = (const float*)d_in[12];
  const float* am1_v  = (const float*)d_in[13];
  const float* ssa_w1 = (const float*)d_in[14];
  const float* ssa_w2 = (const float*)d_in[15];
  const float* ssa_w3 = (const float*)d_in[16];
  const float* ssa_wl = (const float*)d_in[17];
  const float* cs_w1  = (const float*)d_in[18];
  const float* cs_w2  = (const float*)d_in[19];
  const float* cs_w3  = (const float*)d_in[20];
  const float* cs_wl  = (const float*)d_in[21];
  const float* cs_wp  = (const float*)d_in[22];
  const float* cat_w  = (const float*)d_in[23];
  const float* up0_w  = (const float*)d_in[24];
  const float* up0_b  = (const float*)d_in[25];
  const float* up1_w  = (const float*)d_in[26];
  const float* up1_b  = (const float*)d_in[27];

  float* out_x = (float*)d_out;                           // (8,256,128,128)
  float* out_e = (float*)d_out + (size_t)BB * CCH * RSP;  // (8,256,128,128)

  // ---- workspace layout (bump allocator, 256B aligned) ----
  char* wsb = (char*)d_ws;
  size_t off = 0;
  auto alloc = [&](size_t bytes) -> void* {
    void* p = wsb + off;
    off = (off + bytes + 255) & ~(size_t)255;
    return p;
  };
  const size_t CN = (size_t)CCH * NSP;   // 262144
  const size_t IN = (size_t)ICH * NSP;   // 131072
  const size_t W3 = (size_t)9 * ICH * CCH;

  __bf16* wAm0 = (__bf16*)alloc(CCH * CCH * 2);
  __bf16* wAm1 = (__bf16*)alloc(CCH * CCH * 2);
  __bf16* wSl  = (__bf16*)alloc(CCH * ICH * 2);
  __bf16* wCl  = (__bf16*)alloc(CCH * ICH * 2);
  __bf16* wCp  = (__bf16*)alloc(CCH * CCH * 2);
  __bf16* wCat = (__bf16*)alloc(CCH * 2 * CCH * 2);
  __bf16* wU0  = (__bf16*)alloc(CCH * CCH * 2);
  __bf16* wU1  = (__bf16*)alloc(CCH * CCH * 2);
  __bf16* wT1  = (__bf16*)alloc(W3 * 2);
  __bf16* wT2  = (__bf16*)alloc(W3 * 2);
  __bf16* wT3  = (__bf16*)alloc(W3 * 2);
  __bf16* wT4  = (__bf16*)alloc(W3 * 2);
  __bf16* wT5  = (__bf16*)alloc(W3 * 2);
  __bf16* wT6  = (__bf16*)alloc(W3 * 2);

  __bf16* px_h    = (__bf16*)alloc(BB * CN * 2);
  __bf16* pe_h    = (__bf16*)alloc(BB * CN * 2);
  float*  xp_f    = (float*) alloc(BB * CN * 4);
  __bf16* xp_h    = (__bf16*)alloc(BB * CN * 2);
  __bf16* ep_h    = (__bf16*)alloc(BB * CN * 2);
  __bf16* ep2_h   = (__bf16*)alloc(BB * CN * 2);
  __bf16* q_s     = (__bf16*)alloc(BB * IN * 2);
  __bf16* k_s     = (__bf16*)alloc(BB * IN * 2);
  __bf16* v_s     = (__bf16*)alloc(BB * IN * 2);
  __bf16* q_c     = (__bf16*)alloc(BB * IN * 2);
  __bf16* k_c     = (__bf16*)alloc(BB * IN * 2);
  __bf16* v_c     = (__bf16*)alloc(BB * IN * 2);
  float*  Sf      = (float*) alloc((size_t)BB * NSP * NSP * 4);  // 32 MB (reused)
  __bf16* att_h   = (__bf16*)alloc((size_t)BB * NSP * NSP * 2);  // 16 MB (reused)
  __bf16* attout  = (__bf16*)alloc(BB * IN * 2);                 // (reused)
  __bf16* cat_h   = (__bf16*)alloc((size_t)BB * 2 * CN * 2);     // [b][512][1024]
  __bf16* fused_h = (__bf16*)alloc(BB * CN * 2);
  float*  tmp0_f  = (float*) alloc(BB * CN * 4);
  float*  tmp1_f  = (float*) alloc(BB * CN * 4);

  // ---- 1) weight conversions / repacks ----
  auto cvt = [&](const float* s, __bf16* d, int n) {
    cvt_bf16_kernel<<<(n + 255) / 256, 256, 0, stream>>>(s, d, n);
  };
  cvt(am0_w, wAm0, CCH * CCH);
  cvt(am1_w, wAm1, CCH * CCH);
  cvt(ssa_wl, wSl, CCH * ICH);
  cvt(cs_wl, wCl, CCH * ICH);
  cvt(cs_wp, wCp, CCH * CCH);
  cvt(cat_w, wCat, CCH * 2 * CCH);
  cvt(up0_w, wU0, CCH * CCH);
  cvt(up1_w, wU1, CCH * CCH);
  int rblk = (int)((W3 + 255) / 256);
  repack3_kernel<<<rblk, 256, 0, stream>>>(ssa_w1, wT1);
  repack3_kernel<<<rblk, 256, 0, stream>>>(ssa_w2, wT2);
  repack3_kernel<<<rblk, 256, 0, stream>>>(ssa_w3, wT3);
  repack3_kernel<<<rblk, 256, 0, stream>>>(cs_w1, wT4);
  repack3_kernel<<<rblk, 256, 0, stream>>>(cs_w2, wT5);
  repack3_kernel<<<rblk, 256, 0, stream>>>(cs_w3, wT6);

  // ---- 2) pool 128x128 -> 32x32 ----
  pool4_kernel<<<(int)((BB * CN) / 256), 256, 0, stream>>>(x, px_h);
  pool4_kernel<<<(int)((BB * CN) / 256), 256, 0, stream>>>(extra, pe_h);

  dim3 blk(32, 4);
  // ---- 3) xp = bn_relu(conv1x1(px) + b); ep = bn(conv1x1(pe) + b), no relu ----
  gemm_bf16_kernel<<<dim3(CCH / 64, NSP / 64, BB), blk, 0, stream>>>(
      wAm0, 0, CCH, 1, px_h, (long long)CN, NSP,
      xp_f, xp_h, (long long)CN, NSP, CCH,
      am0_b, am0_g, am0_be, am0_m, am0_v, nullptr, 0, 1);
  gemm_bf16_kernel<<<dim3(CCH / 64, NSP / 64, BB), blk, 0, stream>>>(
      wAm1, 0, CCH, 1, pe_h, (long long)CN, NSP,
      nullptr, ep_h, (long long)CN, NSP, CCH,
      am1_b, am1_g, am1_be, am1_m, am1_v, nullptr, 0, 0);

  // ---- 4) ep2 = conv1x1(ep, cs_wp) ----
  gemm_bf16_kernel<<<dim3(CCH / 64, NSP / 64, BB), blk, 0, stream>>>(
      wCp, 0, CCH, 1, ep_h, (long long)CN, NSP,
      nullptr, ep2_h, (long long)CN, NSP, CCH,
      nullptr, nullptr, nullptr, nullptr, nullptr, nullptr, 0, 0);

  // ---- 5) six 3x3 convs (implicit GEMM) ----
  dim3 cgrid(ICH / 64, NSP / 64, BB);
  conv3_bf16_kernel<<<cgrid, blk, 0, stream>>>(wT1, xp_h, q_s);
  conv3_bf16_kernel<<<cgrid, blk, 0, stream>>>(wT2, xp_h, k_s);
  conv3_bf16_kernel<<<cgrid, blk, 0, stream>>>(wT3, xp_h, v_s);
  conv3_bf16_kernel<<<cgrid, blk, 0, stream>>>(wT4, xp_h, q_c);
  conv3_bf16_kernel<<<cgrid, blk, 0, stream>>>(wT5, ep2_h, k_c);
  conv3_bf16_kernel<<<cgrid, blk, 0, stream>>>(wT6, ep2_h, v_c);

  // ---- 6) two attention passes (self, then cross; reuse Sf/att/attout) ----
  for (int pass = 0; pass < 2; ++pass) {
    const __bf16* q = pass ? q_c : q_s;
    const __bf16* k = pass ? k_c : k_s;
    const __bf16* v = pass ? v_c : v_s;
    const __bf16* wl = pass ? wCl : wSl;
    __bf16* catdst = cat_h + (pass ? (size_t)CCH * NSP : 0);

    // S[b][i][j] = sum_c k[b][c][i] * q[b][c][j]   (A = k^T, m-contiguous)
    gemm_bf16_kernel<<<dim3(NSP / 64, NSP / 64, BB), blk, 0, stream>>>(
        k, (long long)IN, 1, NSP, q, (long long)IN, NSP,
        Sf, nullptr, (long long)NSP * NSP, NSP, ICH,
        nullptr, nullptr, nullptr, nullptr, nullptr, nullptr, 0, 0);

    // softmax over j
    softmax_kernel<<<BB * NSP, 256, 0, stream>>>(Sf, att_h);

    // out[b][c][j] = sum_i v[b][c][i] * att[b][i][j]
    gemm_bf16_kernel<<<dim3(ICH / 64, NSP / 64, BB), blk, 0, stream>>>(
        v, (long long)IN, NSP, 1, att_h, (long long)NSP * NSP, NSP,
        nullptr, attout, (long long)IN, NSP, NSP,
        nullptr, nullptr, nullptr, nullptr, nullptr, nullptr, 0, 0);

    // conv1x1(out, wl) + xp -> half of concat buffer [b][512][1024]
    gemm_bf16_kernel<<<dim3(CCH / 64, NSP / 64, BB), blk, 0, stream>>>(
        wl, 0, ICH, 1, attout, (long long)IN, NSP,
        nullptr, catdst, (long long)(2 * CN), NSP, ICH,
        nullptr, nullptr, nullptr, nullptr, nullptr, xp_f, (long long)CN, 0);
  }

  // ---- 7) fused = conv1x1(concat, cat_w)  (K = 512) ----
  gemm_bf16_kernel<<<dim3(CCH / 64, NSP / 64, BB), blk, 0, stream>>>(
      wCat, 0, 2 * CCH, 1, cat_h, (long long)(2 * CN), NSP,
      nullptr, fused_h, (long long)CN, NSP, 2 * CCH,
      nullptr, nullptr, nullptr, nullptr, nullptr, nullptr, 0, 0);

  // ---- 8) output convs at 32x32 (commuted through nearest upsample) ----
  gemm_bf16_kernel<<<dim3(CCH / 64, NSP / 64, BB), blk, 0, stream>>>(
      wU0, 0, CCH, 1, fused_h, (long long)CN, NSP,
      tmp0_f, nullptr, (long long)CN, NSP, CCH,
      up0_b, nullptr, nullptr, nullptr, nullptr, nullptr, 0, 0);
  gemm_bf16_kernel<<<dim3(CCH / 64, NSP / 64, BB), blk, 0, stream>>>(
      wU1, 0, CCH, 1, ep_h, (long long)CN, NSP,
      tmp1_f, nullptr, (long long)CN, NSP, CCH,
      up1_b, nullptr, nullptr, nullptr, nullptr, nullptr, 0, 0);

  // ---- 9) upsample x4 into d_out ----
  upsample4_kernel<<<(int)((BB * CCH * RSP) / 256), 256, 0, stream>>>(tmp0_f, out_x);
  upsample4_kernel<<<(int)((BB * CCH * RSP) / 256), 256, 0, stream>>>(tmp1_f, out_e);
}
